// FusedLlamaQuantizedMLP_59072980189198
// MI455X (gfx1250) — compile-verified, matrix operand order checked by reading, not run on hardware
//
#include <hip/hip_runtime.h>
#include <hip/hip_bf16.h>

#define HIDDEN 4096
#define INTER  11008
#define TOK    2048   // B*S = 2*1024

typedef __bf16 bf16_t;
typedef __attribute__((ext_vector_type(16))) __bf16 v16bf;
typedef __attribute__((ext_vector_type(8)))  __bf16 v8bf;
typedef __attribute__((ext_vector_type(8)))  float  v8f;

union FragU { v16bf v; v8bf h[2]; };
union B4    { bf16_t b[4]; uint2 u; };

// Load a 16x32 bf16 A/B WMMA fragment row for this lane.
// CDNA5 16-bit A/B layout (wave32): lanes 0..15 hold K=[k0+0..8)+[k0+16..24),
// lanes 16..31 hold K=[k0+8..16)+[k0+24..32), row = lane&15.
__device__ __forceinline__ v16bf load_frag_row(const bf16_t* __restrict__ row,
                                               int k0, int hi) {
  FragU f;
  const bf16_t* p = row + k0 + (hi << 3);
  f.h[0] = *(const v8bf*)(p);        // K chunk 0
  f.h[1] = *(const v8bf*)(p + 16);   // K chunk 1
  return f.v;
}

__device__ __forceinline__ v8f wmma_bf16(v16bf a, v16bf b, v8f c) {
  return __builtin_amdgcn_wmma_f32_16x16x32_bf16(
      /*neg_a=*/false, a, /*neg_b=*/false, b,
      /*c_mod=*/(short)0, c, /*reuse_a=*/false, /*reuse_b=*/false);
}

// ---------------- conversion kernels ----------------

__global__ void cvt_i32_to_bf16(const int* __restrict__ src,
                                bf16_t* __restrict__ dst, long long n4) {
  long long i = (long long)blockIdx.x * blockDim.x + threadIdx.x;
  long long stride = (long long)gridDim.x * blockDim.x;
  for (; i < n4; i += stride) {
    int4 w = ((const int4*)src)[i];
    B4 o;
    o.b[0] = (bf16_t)(float)w.x;
    o.b[1] = (bf16_t)(float)w.y;
    o.b[2] = (bf16_t)(float)w.z;
    o.b[3] = (bf16_t)(float)w.w;
    ((uint2*)dst)[i] = o.u;
  }
}

__global__ void cvt_f32_to_bf16(const float* __restrict__ src,
                                bf16_t* __restrict__ dst, long long n4) {
  long long i = (long long)blockIdx.x * blockDim.x + threadIdx.x;
  long long stride = (long long)gridDim.x * blockDim.x;
  for (; i < n4; i += stride) {
    float4 w = ((const float4*)src)[i];
    B4 o;
    o.b[0] = (bf16_t)w.x;
    o.b[1] = (bf16_t)w.y;
    o.b[2] = (bf16_t)w.z;
    o.b[3] = (bf16_t)w.w;
    ((uint2*)dst)[i] = o.u;
  }
}

// ---------------- GEMM1: gate & up + SwiGLU -> h (bf16) ----------------
// Block: 256 threads = 8 waves, 4 M-waves x 2 N-waves. Block tile 128x64.
// Wave tile: 32(M) x 32(N), computing gate AND up sharing the A fragments.
__global__ __launch_bounds__(256, 1)
void gemm_gateup(const bf16_t* __restrict__ xbf,
                 const bf16_t* __restrict__ wg,
                 const bf16_t* __restrict__ wu,
                 const float*  __restrict__ sg,
                 const float*  __restrict__ su,
                 bf16_t* __restrict__ hout) {
  const int lane  = threadIdx.x & 31;
  const int wave  = threadIdx.x >> 5;
  const int waveM = wave & 3;
  const int waveN = wave >> 2;
  const int m0 = blockIdx.x * 128 + waveM * 32;
  const int n0 = blockIdx.y * 64  + waveN * 32;
  const int r  = lane & 15;
  const int hi = lane >> 4;

  const bf16_t* arow0 = xbf + (size_t)(m0 + r) * HIDDEN;
  const bf16_t* arow1 = arow0 + (size_t)16 * HIDDEN;
  const bf16_t* grow0 = wg  + (size_t)(n0 + r) * HIDDEN;
  const bf16_t* grow1 = grow0 + (size_t)16 * HIDDEN;
  const bf16_t* urow0 = wu  + (size_t)(n0 + r) * HIDDEN;
  const bf16_t* urow1 = urow0 + (size_t)16 * HIDDEN;

  v8f ag[2][2], au[2][2];
#pragma unroll
  for (int i = 0; i < 2; ++i)
#pragma unroll
    for (int j = 0; j < 2; ++j) { ag[i][j] = (v8f)0.0f; au[i][j] = (v8f)0.0f; }

  for (int k0 = 0; k0 < HIDDEN; k0 += 32) {
    // Unconditional stream-ahead prefetch: past-the-row addresses land in the
    // adjacent contiguous workspace region (or are speculative & dropped).
    __builtin_prefetch(grow0 + k0 + 256, 0, 1);
    __builtin_prefetch(urow0 + k0 + 256, 0, 1);

    v16bf a0 = load_frag_row(arow0, k0, hi);
    v16bf a1 = load_frag_row(arow1, k0, hi);
    v16bf g0 = load_frag_row(grow0, k0, hi);
    v16bf g1 = load_frag_row(grow1, k0, hi);
    v16bf u0 = load_frag_row(urow0, k0, hi);
    v16bf u1 = load_frag_row(urow1, k0, hi);

    ag[0][0] = wmma_bf16(a0, g0, ag[0][0]);
    ag[0][1] = wmma_bf16(a0, g1, ag[0][1]);
    ag[1][0] = wmma_bf16(a1, g0, ag[1][0]);
    ag[1][1] = wmma_bf16(a1, g1, ag[1][1]);
    au[0][0] = wmma_bf16(a0, u0, au[0][0]);
    au[0][1] = wmma_bf16(a0, u1, au[0][1]);
    au[1][0] = wmma_bf16(a1, u0, au[1][0]);
    au[1][1] = wmma_bf16(a1, u1, au[1][1]);
  }

  // C/D layout: col = n + (lane&15); element e -> row = m + e + 8*(lane>=16)
#pragma unroll
  for (int ni = 0; ni < 2; ++ni) {
    const int col = n0 + ni * 16 + r;
    const float sgv = sg[col];
    const float suv = su[col];
#pragma unroll
    for (int mi = 0; mi < 2; ++mi) {
#pragma unroll
      for (int e = 0; e < 8; ++e) {
        const int row = m0 + mi * 16 + e + 8 * hi;
        float g = ag[mi][ni][e] * sgv;
        float u = au[mi][ni][e] * suv;
        float hval = (g / (1.0f + __expf(-g))) * u;   // silu(g) * u
        hout[(size_t)row * INTER + col] = (bf16_t)hval;
      }
    }
  }
}

// ---------------- GEMM2: out = h @ w_down^T * s_down (fp32 out) ----------------
// Block: 8 waves, 2 M-waves x 4 N-waves. Block tile 64x256. Wave tile 32x64.
__global__ __launch_bounds__(256, 1)
void gemm_down(const bf16_t* __restrict__ hbf,
               const bf16_t* __restrict__ wd,
               const float*  __restrict__ sd,
               float* __restrict__ out) {
  const int lane  = threadIdx.x & 31;
  const int wave  = threadIdx.x >> 5;
  const int waveM = wave & 1;
  const int waveN = wave >> 1;
  const int m0 = blockIdx.x * 64  + waveM * 32;
  const int n0 = blockIdx.y * 256 + waveN * 64;
  const int r  = lane & 15;
  const int hi = lane >> 4;

  const bf16_t* arow0 = hbf + (size_t)(m0 + r) * INTER;
  const bf16_t* arow1 = arow0 + (size_t)16 * INTER;
  const bf16_t* brow[4];
#pragma unroll
  for (int j = 0; j < 4; ++j)
    brow[j] = wd + (size_t)(n0 + j * 16 + r) * INTER;

  v8f acc[2][4];
#pragma unroll
  for (int i = 0; i < 2; ++i)
#pragma unroll
    for (int j = 0; j < 4; ++j) acc[i][j] = (v8f)0.0f;

  for (int k0 = 0; k0 < INTER; k0 += 32) {
    __builtin_prefetch(brow[0] + k0 + 256, 0, 1);
    __builtin_prefetch(brow[2] + k0 + 256, 0, 1);

    v16bf a0 = load_frag_row(arow0, k0, hi);
    v16bf a1 = load_frag_row(arow1, k0, hi);
#pragma unroll
    for (int j = 0; j < 4; ++j) {
      v16bf b = load_frag_row(brow[j], k0, hi);
      acc[0][j] = wmma_bf16(a0, b, acc[0][j]);
      acc[1][j] = wmma_bf16(a1, b, acc[1][j]);
    }
  }

#pragma unroll
  for (int j = 0; j < 4; ++j) {
    const int col = n0 + j * 16 + r;
    const float sv = sd[col];
#pragma unroll
    for (int i = 0; i < 2; ++i) {
#pragma unroll
      for (int e = 0; e < 8; ++e) {
        const int row = m0 + i * 16 + e + 8 * hi;
        out[(size_t)row * HIDDEN + col] = acc[i][j][e] * sv;
      }
    }
  }
}

// ---------------- launch ----------------

extern "C" void kernel_launch(void* const* d_in, const int* in_sizes, int n_in,
                              void* d_out, int out_size, void* d_ws, size_t ws_size,
                              hipStream_t stream) {
  const float* x      = (const float*)d_in[0];
  const int*   w_gate = (const int*)d_in[1];
  const float* s_gate = (const float*)d_in[2];
  const int*   w_up   = (const int*)d_in[3];
  const float* s_up   = (const float*)d_in[4];
  const int*   w_down = (const int*)d_in[5];
  const float* s_down = (const float*)d_in[6];
  float* out = (float*)d_out;

  const size_t IH = (size_t)INTER * HIDDEN;      // 45,088,768
  const size_t TH = (size_t)TOK * HIDDEN;        //  8,388,608
  const size_t TI = (size_t)TOK * INTER;         // 22,544,384

  char* ws = (char*)d_ws;
  bf16_t* wg_bf = (bf16_t*)ws;              ws += IH * sizeof(bf16_t);
  bf16_t* wu_bf = (bf16_t*)ws;              ws += IH * sizeof(bf16_t);
  bf16_t* wd_bf = (bf16_t*)ws;              ws += IH * sizeof(bf16_t);
  bf16_t* x_bf  = (bf16_t*)ws;              ws += TH * sizeof(bf16_t);
  bf16_t* h_bf  = (bf16_t*)ws;              ws += TI * sizeof(bf16_t);

  // 1) dequant-widen weights int32 -> bf16 (values are exact in bf16);
  //    bf16 copies (180 MB gate+up) stay L2-resident for the GEMM passes.
  cvt_i32_to_bf16<<<4096, 256, 0, stream>>>(w_gate, wg_bf, (long long)(IH / 4));
  cvt_i32_to_bf16<<<4096, 256, 0, stream>>>(w_up,   wu_bf, (long long)(IH / 4));
  cvt_i32_to_bf16<<<4096, 256, 0, stream>>>(w_down, wd_bf, (long long)(IH / 4));
  // 2) activations fp32 -> bf16
  cvt_f32_to_bf16<<<2048, 256, 0, stream>>>(x, x_bf, (long long)(TH / 4));

  // 3) gate/up GEMM + per-channel scales + SwiGLU -> h (bf16)
  //    grid: M = 2048/128 = 16 (fast axis, shares weight tiles via L2),
  //          N = 11008/64 = 172
  gemm_gateup<<<dim3(16, 172), 256, 0, stream>>>(x_bf, wg_bf, wu_bf,
                                                 s_gate, s_up, h_bf);

  // 4) down GEMM + scale -> fp32 out
  //    grid: M = 2048/64 = 32, N = 4096/256 = 16
  gemm_down<<<dim3(32, 16), 256, 0, stream>>>(h_bf, wd_bf, s_down, out);
}
